// GNNEncoder_37864431681686
// MI455X (gfx1250) — compile-verified, matrix operand order checked by reading, not run on hardware
//
#include <hip/hip_runtime.h>

#define NN 100000
#define NE 800000
#define HID 128
#define NL 3
#define NG 64

typedef __attribute__((ext_vector_type(16))) __bf16 v16bf;
typedef __attribute__((ext_vector_type(8)))  __bf16 v8bf;
typedef __attribute__((ext_vector_type(8)))  float  v8f;
typedef __attribute__((ext_vector_type(4)))  float  f4;

// ---------------------------------------------------------------------------
// BF16 operand fragment helpers (CDNA5 WMMA 16-bit layout, CDNA5 ISA):
// lane l = (p | hi<<4): holds K in [8*hi, 8*hi+8) and [16+8*hi, 16+8*hi+8)
// for row/col p. Two contiguous 8-element (16B) chunks per 32-wide K step.
// ---------------------------------------------------------------------------
__device__ __forceinline__ v16bf load_a_frag(const float* __restrict__ A,
                                             int row, int kk, int hi) {
  const float* ap = A + row * HID + kk + 8 * hi;
  f4 f0 = *(const f4*)(ap);
  f4 f1 = *(const f4*)(ap + 4);
  f4 f2 = *(const f4*)(ap + 16);
  f4 f3 = *(const f4*)(ap + 20);
  v16bf a;
  a[0]  = (__bf16)f0[0]; a[1]  = (__bf16)f0[1]; a[2]  = (__bf16)f0[2]; a[3]  = (__bf16)f0[3];
  a[4]  = (__bf16)f1[0]; a[5]  = (__bf16)f1[1]; a[6]  = (__bf16)f1[2]; a[7]  = (__bf16)f1[3];
  a[8]  = (__bf16)f2[0]; a[9]  = (__bf16)f2[1]; a[10] = (__bf16)f2[2]; a[11] = (__bf16)f2[3];
  a[12] = (__bf16)f3[0]; a[13] = (__bf16)f3[1]; a[14] = (__bf16)f3[2]; a[15] = (__bf16)f3[3];
  return a;
}

__device__ __forceinline__ v16bf load_b_frag(const __bf16* __restrict__ wt,
                                             int col, int kk, int hi) {
  const __bf16* wp = wt + col * HID + kk + 8 * hi;
  v8bf lo = *(const v8bf*)(wp);       // ds_load_b128
  v8bf hi8 = *(const v8bf*)(wp + 16); // ds_load_b128
  return __builtin_shufflevector(lo, hi8, 0, 1, 2, 3, 4, 5, 6, 7,
                                 8, 9, 10, 11, 12, 13, 14, 15);
}

// Convert W (row-major [k][n], fp32) into LDS as BF16 transposed wt[n][k].
__device__ __forceinline__ void fill_wt(const float* __restrict__ W,
                                        __bf16* wt) {
  for (int idx = threadIdx.x; idx < HID * HID; idx += blockDim.x) {
    int k = idx >> 7;
    int n = idx & 127;
    wt[n * HID + k] = (__bf16)W[idx];
  }
  __syncthreads();
}

// One wave: 16 rows x full 128 cols, K = 128 (4 WMMA K-steps x 8 N-tiles).
__device__ __forceinline__ void wave_gemm_tile(const float* __restrict__ A,
                                               int tile, const __bf16* wt,
                                               int lane, v8f acc[8]) {
  const int p  = lane & 15;
  const int hi = lane >> 4;
  const int row = tile * 16 + p;
#pragma unroll
  for (int kb = 0; kb < 4; ++kb) {
    const int kk = kb * 32;
    v16bf a = load_a_frag(A, row, kk, hi);
#pragma unroll
    for (int t = 0; t < 8; ++t) {
      v16bf b = load_b_frag(wt, t * 16 + p, kk, hi);
      acc[t] = __builtin_amdgcn_wmma_f32_16x16x32_bf16(
          false, a, false, b, (short)0, acc[t], false, false);
    }
  }
}

// ---------------------------------------------------------------------------
// C = A[M,128] @ W[128,128] (+ bias). One wave per 16-row tile.
// ---------------------------------------------------------------------------
__global__ void __launch_bounds__(256)
gemm_node(const float* __restrict__ A, const float* __restrict__ W,
          const float* __restrict__ bias, float* __restrict__ C,
          int M_rows, int has_bias) {
  __shared__ __bf16 wt[HID * HID];
  fill_wt(W, wt);
  const int lane = threadIdx.x & 31;
  const int wave = threadIdx.x >> 5;
  const int tile = blockIdx.x * 8 + wave;
  const int numTiles = M_rows >> 4;
  if (tile >= numTiles) return;

  v8f acc[8];
#pragma unroll
  for (int t = 0; t < 8; ++t)
#pragma unroll
    for (int r = 0; r < 8; ++r) acc[t][r] = 0.0f;

  wave_gemm_tile(A, tile, wt, lane, acc);

  const int n = lane & 15;
  const int hi = lane >> 4;
#pragma unroll
  for (int t = 0; t < 8; ++t) {
    const int col = t * 16 + n;
    const float bc = has_bias ? bias[col] : 0.0f;
#pragma unroll
    for (int r = 0; r < 8; ++r) {
      const int row = tile * 16 + 8 * hi + r;  // C layout: VGPR r -> M = r+8*hi
      C[row * HID + col] = acc[t][r] + bc;
    }
  }
}

// ---------------------------------------------------------------------------
// Fused: ew = edge_attr @ W ; msgs = h[src]*ew ; agg[dst] += msgs (atomics).
// One wave per 16-edge tile. h/agg are L2-resident (51 MB << 192 MB L2).
// ---------------------------------------------------------------------------
__global__ void __launch_bounds__(256)
edge_gate_scatter(const float* __restrict__ EA, const float* __restrict__ W,
                  const int* __restrict__ src, const int* __restrict__ dst,
                  const float* __restrict__ h, float* __restrict__ agg) {
  __shared__ __bf16 wt[HID * HID];
  fill_wt(W, wt);
  const int lane = threadIdx.x & 31;
  const int wave = threadIdx.x >> 5;
  const int tile = blockIdx.x * 8 + wave;
  if (tile >= (NE >> 4)) return;

  v8f acc[8];
#pragma unroll
  for (int t = 0; t < 8; ++t)
#pragma unroll
    for (int r = 0; r < 8; ++r) acc[t][r] = 0.0f;

  wave_gemm_tile(EA, tile, wt, lane, acc);

  const int n = lane & 15;
  const int hi = lane >> 4;
#pragma unroll
  for (int r = 0; r < 8; ++r) {
    const int e = tile * 16 + 8 * hi + r;  // edge index for accumulator row r
    const int s = src[e];
    const int d = dst[e];
    const float* hrow = h + s * HID;
    float* arow = agg + d * HID;
#pragma unroll
    for (int t = 0; t < 8; ++t) {
      const int col = t * 16 + n;
      atomicAdd(arow + col, acc[t][r] * hrow[col]);  // global_atomic_add_f32
    }
  }
}

// ---------------------------------------------------------------------------
// Per node row: h = relu(agg + b); (+residual); x = layernorm(h)*gamma+beta.
// One wave per row (128 cols -> 4 per lane), wave32 shuffle reductions.
// ---------------------------------------------------------------------------
__global__ void __launch_bounds__(256)
post_ln(const float* __restrict__ agg, const float* __restrict__ bias,
        const float* __restrict__ gamma, const float* __restrict__ beta,
        float* __restrict__ x, int addResidual) {
  const int lane = threadIdx.x & 31;
  const int wave = threadIdx.x >> 5;
  const int row = blockIdx.x * 8 + wave;
  if (row >= NN) return;
  const int c = lane * 4;

  f4 v = *(const f4*)(agg + row * HID + c);
  f4 b = *(const f4*)(bias + c);
  f4 hv;
#pragma unroll
  for (int j = 0; j < 4; ++j) hv[j] = fmaxf(v[j] + b[j], 0.0f);
  if (addResidual) {
    f4 rv = *(const f4*)(x + row * HID + c);
#pragma unroll
    for (int j = 0; j < 4; ++j) hv[j] += rv[j];
  }

  float s = hv[0] + hv[1] + hv[2] + hv[3];
#pragma unroll
  for (int off = 16; off >= 1; off >>= 1) s += __shfl_xor(s, off, 32);
  const float mean = s * (1.0f / 128.0f);

  f4 dv;
  float q = 0.0f;
#pragma unroll
  for (int j = 0; j < 4; ++j) { dv[j] = hv[j] - mean; q += dv[j] * dv[j]; }
#pragma unroll
  for (int off = 16; off >= 1; off >>= 1) q += __shfl_xor(q, off, 32);
  const float inv = rsqrtf(q * (1.0f / 128.0f) + 1e-5f);

  f4 g = *(const f4*)(gamma + c);
  f4 be = *(const f4*)(beta + c);
  f4 o;
#pragma unroll
  for (int j = 0; j < 4; ++j) o[j] = dv[j] * inv * g[j] + be[j];
  *(f4*)(x + row * HID + c) = o;
}

// ---------------------------------------------------------------------------
// Global mean pool: sums[batch[n]] += y[n]; counts[batch[n]] += 1.
// ---------------------------------------------------------------------------
__global__ void __launch_bounds__(256)
pool_accum(const float* __restrict__ y, const int* __restrict__ batch,
           float* __restrict__ sums, float* __restrict__ counts) {
  const int lane = threadIdx.x & 31;
  const int wave = threadIdx.x >> 5;
  const int row = blockIdx.x * 8 + wave;
  if (row >= NN) return;
  const int g = batch[row];
  const int c = lane * 4;
  f4 v = *(const f4*)(y + row * HID + c);
  atomicAdd(sums + g * HID + c + 0, v[0]);
  atomicAdd(sums + g * HID + c + 1, v[1]);
  atomicAdd(sums + g * HID + c + 2, v[2]);
  atomicAdd(sums + g * HID + c + 3, v[3]);
  if (lane == 0) atomicAdd(counts + g, 1.0f);
}

__global__ void __launch_bounds__(256)
pool_final(const float* __restrict__ sums, const float* __restrict__ counts,
           float* __restrict__ out) {
  const int idx = blockIdx.x * blockDim.x + threadIdx.x;
  if (idx >= NG * HID) return;
  const int g = idx >> 7;
  out[idx] = sums[idx] / fmaxf(counts[g], 1.0f);
}

// ---------------------------------------------------------------------------
extern "C" void kernel_launch(void* const* d_in, const int* in_sizes, int n_in,
                              void* d_out, int out_size, void* d_ws,
                              size_t ws_size, hipStream_t stream) {
  const float* node_features = (const float*)d_in[0];
  const float* edge_attr     = (const float*)d_in[1];
  const float* in_w          = (const float*)d_in[2];
  const float* in_b          = (const float*)d_in[3];
  const float* gcn_w         = (const float*)d_in[4];
  const float* gcn_ew        = (const float*)d_in[5];
  const float* gcn_b         = (const float*)d_in[6];
  const float* ln_s          = (const float*)d_in[7];
  const float* ln_b          = (const float*)d_in[8];
  const float* out_w         = (const float*)d_in[9];
  const float* out_b         = (const float*)d_in[10];
  const int*   ei            = (const int*)d_in[11];
  const int*   batch         = (const int*)d_in[12];
  float* out = (float*)d_out;

  char* ws = (char*)d_ws;
  const size_t nh = (size_t)NN * HID * sizeof(float);
  float* x      = (float*)(ws);
  float* h      = (float*)(ws + nh);
  float* agg    = (float*)(ws + 2 * nh);
  float* sums   = (float*)(ws + 3 * nh);
  float* counts = sums + NG * HID;

  const int* src = ei;
  const int* dst = ei + NE;

  const dim3 blk(256);
  const int nodeTiles  = NN / 16;               // 6250
  const int nodeBlocks = (nodeTiles + 7) / 8;   // 782
  const int edgeBlocks = (NE / 16) / 8;         // 6250 (exact)
  const int rowBlocks  = NN / 8;                // 12500 (exact)

  // x = node_features @ in_w + in_b
  gemm_node<<<nodeBlocks, blk, 0, stream>>>(node_features, in_w, in_b, x, NN, 1);

  for (int i = 0; i < NL; ++i) {
    // h = x @ gcn_w[i]
    gemm_node<<<nodeBlocks, blk, 0, stream>>>(x, gcn_w + i * HID * HID,
                                              nullptr, h, NN, 0);
    hipMemsetAsync(agg, 0, nh, stream);
    // agg[dst] += (edge_attr @ gcn_ew[i]) * h[src]
    edge_gate_scatter<<<edgeBlocks, blk, 0, stream>>>(
        edge_attr, gcn_ew + i * HID * HID, src, dst, h, agg);
    // x = LN(relu(agg + b) [+ x])
    post_ln<<<rowBlocks, blk, 0, stream>>>(agg, gcn_b + i * HID, ln_s + i * HID,
                                           ln_b + i * HID, x, (i > 0) ? 1 : 0);
  }

  // y = x @ out_w + out_b  (into h)
  gemm_node<<<nodeBlocks, blk, 0, stream>>>(x, out_w, out_b, h, NN, 1);

  hipMemsetAsync(sums, 0, (size_t)(NG * HID + NG) * sizeof(float), stream);
  pool_accum<<<rowBlocks, blk, 0, stream>>>(h, batch, sums, counts);
  pool_final<<<(NG * HID + 255) / 256, 256, 0, stream>>>(sums, counts, out);
}
